// ShoppingLSTM_51668456571242
// MI455X (gfx1250) — matrix operand order repaired and checked by reading, not compiled, not run on hardware
//
#include <hip/hip_runtime.h>
#include <hip/hip_bf16.h>
#include <math.h>

// Problem constants (match reference)
#define EMB_D   256
#define H_D     256
#define OUT_D   10
#define B_D     128
#define S_D     256
#define G4H     1024   // 4*H

typedef __attribute__((ext_vector_type(16))) __bf16 v16bf;
typedef __attribute__((ext_vector_type(8)))  float  v8f;

union BFrag {            // 16x32 / 32x16 bf16 WMMA operand fragment (8 VGPRs)
    v16bf v;
    uint4 q[2];
};
union FFrag {            // 16x16 f32 accumulator fragment (8 VGPRs)
    v8f   v;
    float f[8];
    uint4 q[2];
};

__device__ __forceinline__ unsigned short f2bf(float x) {
    unsigned u = __builtin_bit_cast(unsigned, x);
    unsigned r = u + 0x7FFFu + ((u >> 16) & 1u);   // round-to-nearest-even
    return (unsigned short)(r >> 16);
}
__device__ __forceinline__ float bf2f(unsigned short b) {
    return __builtin_bit_cast(float, ((unsigned)b) << 16);
}

// gfx1250 has a hardware v_tanh_f32 transcendental; use it when clang exposes it.
#if __has_builtin(__builtin_amdgcn_tanhf)
__device__ __forceinline__ float tanh_(float x) { return __builtin_amdgcn_tanhf(x); }
__device__ __forceinline__ float sigmoid_(float x) {
    return 0.5f * (1.0f + __builtin_amdgcn_tanhf(0.5f * x));
}
#else
__device__ __forceinline__ float tanh_(float x) { return tanhf(x); }
__device__ __forceinline__ float sigmoid_(float x) {
    return 1.0f / (1.0f + __expf(-x));
}
#endif

// ---------------------------------------------------------------------------
// Pack a [256, 1024] f32 weight matrix into per-lane bf16 B-fragment layout.
// Tile (ntile 0..63, kt 0..7): lane L holds column N = ntile*16 + (L&15),
// K = kt*32 + (L>>4)*16 + j, j = 0..15 contiguous  (contiguous k order; A loads use the same).
// dst element index = ((ntile*8 + kt)*32 + lane)*16 + j
// ---------------------------------------------------------------------------
__global__ void pack_b_kernel(const float* __restrict__ src,
                              unsigned short* __restrict__ dst) {
    int i = blockIdx.x * blockDim.x + threadIdx.x;   // 0 .. 262143
    int tIdx = i >> 9;            // tile index (64*8)
    int rem  = i & 511;
    int lane = rem >> 4;
    int j    = rem & 15;
    int ntile = tIdx >> 3;
    int kt    = tIdx & 7;
    int K = kt * 32 + (lane >> 4) * 16 + j;
    int N = ntile * 16 + (lane & 15);
    dst[i] = f2bf(src[K * G4H + N]);
}

// ---------------------------------------------------------------------------
// Embedding gather: seq[row, e] = bf16(table[x[row], e]),  row = b*S + s
// ---------------------------------------------------------------------------
__global__ void embed_kernel(const int* __restrict__ x,
                             const float* __restrict__ tab,
                             unsigned short* __restrict__ seq) {
    long long i = (long long)blockIdx.x * blockDim.x + threadIdx.x;
    int row = (int)(i >> 8);
    int e   = (int)(i & 255);
    int tok = x[row];
    seq[i] = f2bf(tab[(long long)tok * EMB_D + e]);
}

// ---------------------------------------------------------------------------
// One LSTM layer, persistent grid: 16 workgroups x 8 waves = 128 waves.
// Wave (m = wave, n0 = blockIdx*16) owns the 16x16 hidden block
// rows [16m,16m+16) x cols [n0,n0+16) and its four gate tiles.
// Cell state c stays in a v8f register fragment across all 256 steps.
// Per step: 64 x v_wmma_f32_16x16x32_bf16. Device-wide atomic barrier
// publishes h between steps. W fragments are loop-invariant -> the compiler
// keeps them register-resident (gfx1250 high-VGPR bank); U comes from LDS.
// ---------------------------------------------------------------------------
__launch_bounds__(256, 1)
__global__ void lstm_layer_kernel(const unsigned short* __restrict__ seq_in,
                                  unsigned short* __restrict__ seq_out,
                                  const unsigned short* __restrict__ Wp,
                                  const unsigned short* __restrict__ Up,
                                  const float* __restrict__ bias,
                                  unsigned short* __restrict__ h_buf,
                                  float* __restrict__ c_buf,
                                  unsigned* bar) {
    __shared__ unsigned short ldsU[4 * 8 * 512];   // 32 KB: U slice for this n0
    const int tid    = threadIdx.x;
    const int n0tile = blockIdx.x;                 // 0..15
    const int wave   = tid >> 5;                   // 0..7  -> m tile
    const int lane   = tid & 31;

    // Stage this block's U fragments into LDS (32 tiles x 1 KB)
    {
        const uint4* usrc = (const uint4*)Up;
        uint4*       udst = (uint4*)ldsU;
        for (int it = tid; it < 2048; it += 256) {
            int tile = it >> 6;                    // 64 x uint4 per tile
            int off  = it & 63;
            int g    = tile >> 3;
            int kt   = tile & 7;
            int gsrc = (n0tile + 16 * g) * 8 + kt; // global tile index
            udst[tile * 64 + off] = usrc[gsrc * 64 + off];
        }
    }
    __syncthreads();

    const int mtile   = wave;
    const int rowbase = mtile * 16;
    const int nIdx    = lane & 15;
    const int khalf   = lane >> 4;
    const int n0      = n0tile * 16;
    const int arow    = rowbase + nIdx;            // A-fragment source row

    float bg[4];                                   // per-gate bias (per column)
#pragma unroll
    for (int g = 0; g < 4; ++g) bg[g] = bias[g * 256 + n0 + nIdx];

    // Load persistent cell-state fragment (threads across layer launches)
    FFrag c;
    {
        const float* cp = c_buf + (((n0tile * 8 + mtile) * 32 + lane) * 8);
        c.q[0] = *(const uint4*)(cp);
        c.q[1] = *(const uint4*)(cp + 4);
    }

    const unsigned short* xrow = seq_in + (long long)arow * S_D * H_D;
    const unsigned short* hrow = h_buf + arow * H_D;

#pragma unroll 1
    for (int t = 0; t < S_D; ++t) {
        FFrag acc[4];
#pragma unroll
        for (int g = 0; g < 4; ++g)
#pragma unroll
            for (int e = 0; e < 8; ++e) acc[g].f[e] = bg[g];

        const unsigned short* xt = xrow + (long long)t * H_D;
        if (t + 1 < S_D) __builtin_prefetch(xt + H_D, 0, 1);  // global_prefetch

#pragma unroll
        for (int kt = 0; kt < 8; ++kt) {
            const int colbase = kt * 32 + khalf * 16;          // contiguous k = 16h + s (matches packed B)
            BFrag ax, ah;
            ax.q[0] = *(const uint4*)(xt + colbase);
            ax.q[1] = *(const uint4*)(xt + colbase + 8);
            ah.q[0] = *(const uint4*)(hrow + colbase);
            ah.q[1] = *(const uint4*)(hrow + colbase + 8);

            // Batch ALL operand loads for this k-tile into distinct registers
            // first, so one wait covers them and WMMAs overlap the DS latency.
            BFrag wf[4], uf[4];
#pragma unroll
            for (int g = 0; g < 4; ++g) {
                const unsigned short* wp =
                    Wp + (((n0tile + 16 * g) * 8 + kt) * 512 + lane * 16);
                wf[g].q[0] = *(const uint4*)(wp);
                wf[g].q[1] = *(const uint4*)(wp + 8);
                const unsigned short* up = ldsU + ((g * 8 + kt) * 512 + lane * 16);
                uf[g].q[0] = *(const uint4*)(up);
                uf[g].q[1] = *(const uint4*)(up + 8);
            }
            // x @ W contribution (W fragments are register-resident)
#pragma unroll
            for (int g = 0; g < 4; ++g)
                acc[g].v = __builtin_amdgcn_wmma_f32_16x16x32_bf16(
                    false, ax.v, false, wf[g].v, (short)0, acc[g].v, false, false);
            // h @ U contribution
#pragma unroll
            for (int g = 0; g < 4; ++g)
                acc[g].v = __builtin_amdgcn_wmma_f32_16x16x32_bf16(
                    false, ah.v, false, uf[g].v, (short)0, acc[g].v, false, false);
        }

        // Elementwise LSTM cell (gate order i,f,g,o), write h block
#pragma unroll
        for (int e = 0; e < 8; ++e) {
            float iv = sigmoid_(acc[0].f[e]);
            float fv = sigmoid_(acc[1].f[e]);
            float gv = tanh_(acc[2].f[e]);
            float ov = sigmoid_(acc[3].f[e]);
            float cn = fv * c.f[e] + iv * gv;
            c.f[e]   = cn;
            float hv = ov * tanh_(cn);
            unsigned short hb = f2bf(hv);
            int row = rowbase + khalf * 8 + e;     // C-fragment M mapping
            int col = n0 + nIdx;
            h_buf[row * H_D + col] = hb;
            seq_out[((long long)row * S_D + t) * H_D + col] = hb;
        }

        // Device-wide barrier: publish h for all 16 workgroups
        __syncthreads();
        if (tid == 0) {
            __hip_atomic_fetch_add(bar, 1u, __ATOMIC_RELEASE,
                                   __HIP_MEMORY_SCOPE_AGENT);
            unsigned target = (unsigned)(t + 1) * gridDim.x;
            while (__hip_atomic_load(bar, __ATOMIC_ACQUIRE,
                                     __HIP_MEMORY_SCOPE_AGENT) < target) {
                __builtin_amdgcn_s_sleep(1);
            }
        }
        __syncthreads();
    }

    // Persist cell state for the next layer launch
    {
        float* cp = c_buf + (((n0tile * 8 + mtile) * 32 + lane) * 8);
        *(uint4*)(cp)     = c.q[0];
        *(uint4*)(cp + 4) = c.q[1];
    }
}

// ---------------------------------------------------------------------------
// Final FC: out[b,o] = sum_n h[b,n] * fcW[n,o] + fcb[o]   (tiny, 1280 outputs)
// ---------------------------------------------------------------------------
__global__ void fc_kernel(const unsigned short* __restrict__ h,
                          const float* __restrict__ fcW,
                          const float* __restrict__ fcb,
                          float* __restrict__ out) {
    int i = blockIdx.x * blockDim.x + threadIdx.x;
    if (i >= B_D * OUT_D) return;
    int b = i / OUT_D, o = i % OUT_D;
    float s = fcb[o];
    for (int n = 0; n < H_D; ++n) s += bf2f(h[b * H_D + n]) * fcW[n * OUT_D + o];
    out[i] = s;
}

// ---------------------------------------------------------------------------
extern "C" void kernel_launch(void* const* d_in, const int* in_sizes, int n_in,
                              void* d_out, int out_size, void* d_ws, size_t ws_size,
                              hipStream_t stream) {
    (void)in_sizes; (void)n_in; (void)out_size; (void)ws_size;
    const int*   x   = (const int*)  d_in[0];
    const float* tab = (const float*)d_in[1];
    const float* W0  = (const float*)d_in[2];
    const float* U0  = (const float*)d_in[3];
    const float* b0  = (const float*)d_in[4];
    const float* Ws  = (const float*)d_in[5];
    const float* Us  = (const float*)d_in[6];
    const float* bs  = (const float*)d_in[7];
    const float* fcW = (const float*)d_in[8];
    const float* fcb = (const float*)d_in[9];
    float* out = (float*)d_out;

    char* ws = (char*)d_ws;
    size_t off = 0;
    auto alloc = [&](size_t bytes) -> void* {
        void* p = ws + off;
        off = (off + bytes + 255) & ~(size_t)255;
        return p;
    };
    const size_t seq_bytes = (size_t)B_D * S_D * H_D * 2;       // 16.8 MB bf16
    unsigned short* seqA = (unsigned short*)alloc(seq_bytes);
    unsigned short* seqB = (unsigned short*)alloc(seq_bytes);
    unsigned short* h_buf = (unsigned short*)alloc((size_t)B_D * H_D * 2);
    float*          c_buf = (float*)alloc((size_t)B_D * H_D * 4);
    unsigned short* W0p = (unsigned short*)alloc((size_t)256 * G4H * 2);
    unsigned short* U0p = (unsigned short*)alloc((size_t)256 * G4H * 2);
    unsigned short* Wsp = (unsigned short*)alloc((size_t)256 * G4H * 2);
    unsigned short* Usp = (unsigned short*)alloc((size_t)256 * G4H * 2);
    unsigned* bar = (unsigned*)alloc(256);

    // Pack weights into WMMA B-fragment layout (bf16)
    pack_b_kernel<<<1024, 256, 0, stream>>>(W0, W0p);
    pack_b_kernel<<<1024, 256, 0, stream>>>(U0, U0p);
    pack_b_kernel<<<1024, 256, 0, stream>>>(Ws, Wsp);
    pack_b_kernel<<<1024, 256, 0, stream>>>(Us, Usp);

    // Embedding gather (f32 -> bf16)
    embed_kernel<<<(B_D * S_D * H_D) / 256, 256, 0, stream>>>(x, tab, seqA);

    // Initial state h0 = c0 = 0
    hipMemsetAsync(h_buf, 0, (size_t)B_D * H_D * 2, stream);
    hipMemsetAsync(c_buf, 0, (size_t)B_D * H_D * 4, stream);

    // 12 layers: state (h,c) threads through; layers 2..12 share weights
    for (int l = 0; l < 12; ++l) {
        hipMemsetAsync(bar, 0, sizeof(unsigned), stream);
        const unsigned short* sin  = (l & 1) ? seqB : seqA;
        unsigned short*       sout = (l & 1) ? seqA : seqB;
        lstm_layer_kernel<<<16, 256, 0, stream>>>(
            sin, sout,
            (l == 0) ? W0p : Wsp,
            (l == 0) ? U0p : Usp,
            (l == 0) ? b0  : bs,
            h_buf, c_buf, bar);
    }

    // Final projection from last hidden state
    fc_kernel<<<(B_D * OUT_D + 255) / 256, 256, 0, stream>>>(h_buf, fcW, fcb, out);
}
